// GPT_88055419502967
// MI455X (gfx1250) — compile-verified, matrix operand order checked by reading, not run on hardware
//
#include <hip/hip_runtime.h>
#include <hip/hip_bf16.h>
#include <math.h>

// ============================================================================
// CLIP-style dual transformer encoder for MI455X (gfx1250, wave32, WMMA).
// v4: - QKV fused into one 4096x2304 GEMM; V-segment written transposed
//       (VT[b,h,d,t]) in the epilogue so flash attention's P.V B-fragments
//       become contiguous b128 loads (no strided u16 gathers)
//     - fused flash attention (scores never touch HBM)
//     - 128x128 macro-tile GEMM, 32x64 wave tiles, double-buffered 40KB LDS
//     - weights f16-converted once per branch; residual stream f32
//
// d_in leaf order assumption: pytree leaves in dict-insertion order:
//   0 formula_seq  1 formula_padding  2 points  3 points_padding
//   4..19  f_params: ln1_s, ln1_b, ln2_s, ln2_b, Wq, bq, Wk, bk, Wv, bv,
//                    Wo, bo, Wm1, bm1, Wm2, bm2   (leading dim L=4)
//   20..35 p_params (same order)
//   36..43 emb: tok_w, tok_b, p1_w, p1_b, p2_w, p2_b, p3_w, p3_b
//   44..56 head: lnf_s, lnf_b, lnp_s, lnp_b, pf_W1, pf_b1, pf_W2, pf_b2,
//                pp_W1, pp_b1, pp_W2, pp_b2, logit_scale
// ============================================================================

typedef __attribute__((ext_vector_type(16))) _Float16 v16h;
typedef __attribute__((ext_vector_type(4)))  _Float16 v4h;
typedef __attribute__((ext_vector_type(8)))  float    v8f;

#define DIMD   768
#define NHEAD  12
#define HD     64
#define NBATCH 16
#define SEQ    256
#define NLAYER 4
#define MROWS  (NBATCH * SEQ)   // 4096
#define FFDIM  (4 * DIMD)       // 3072
#define QKVN   (3 * DIMD)       // 2304
#define LN_EPS 1e-5f

#define MBLK   128
#define NBLK   128
#define KSTEP  32
#define LDSTR  40               // LDS row stride in halves (32 + 8 pad)

#define WMMA16(A, B, C) \
  __builtin_amdgcn_wmma_f32_16x16x32_f16(false, (A), false, (B), (short)0, (C), false, false)

__device__ __forceinline__ float gelu_exact(float x) {
  return 0.5f * x * (1.0f + erff(x * 0.70710678118654752f));
}

union Frag {
  v16h   v;
  float4 q[2];
};

// ---------------------------------------------------------------------------
// Fragment loaders. CDNA5 16-bit A (16x32): lane = (r, hh); half hh holds
// K in [8*hh, 8*hh+8) and [16+8*hh, 24+8*hh). B mirrors with lanes = N.
// Identical slot->k map for A and B keeps the contraction consistent.
// ---------------------------------------------------------------------------
__device__ __forceinline__ v16h frag_lds(const _Float16* __restrict__ buf, int row, int hh) {
  const _Float16* p = buf + row * LDSTR + hh * 8;
  Frag f;
  f.q[0] = *(const float4*)p;
  f.q[1] = *(const float4*)(p + 16);
  return f.v;
}

__device__ __forceinline__ v16h frag_gbl_h(const _Float16* __restrict__ p) {
  Frag f;
  f.q[0] = *(const float4*)p;
  f.q[1] = *(const float4*)(p + 16);
  return f.v;
}

__device__ __forceinline__ v16h frag_gbl_f32(const float* __restrict__ p) {
  v16h f;
#pragma unroll
  for (int i = 0; i < 8; ++i) {
    f[i]     = (_Float16)p[i];
    f[i + 8] = (_Float16)p[16 + i];
  }
  return f;
}

// ---------------------------------------------------------------------------
// Tiled WMMA GEMM:  C[M,N] = A[M,K](f16) @ W[N,K](f16)^T + bias[N]
// FUSE: 0 = f16 out, 1 = GELU -> f16 out, 2 = +residual(f32) -> f32 out,
//       3 = f32 out, 4 = fused-QKV: cols [0,768)->Qo, [768,1536)->Ko,
//           [1536,2304)->Vt transposed as VT[b,h,d,t].
// 256 thr / 8 waves; macro-tile 128x128; wave tile 32x64 (2x4 WMMA tiles).
// Double-buffered LDS (40KB), 1 barrier per K-step.
// ---------------------------------------------------------------------------
template <int FUSE>
__global__ __launch_bounds__(256)
void wmma_gemm_tiled(const _Float16* __restrict__ A, const _Float16* __restrict__ W,
                     const float* __restrict__ bias, const float* __restrict__ R,
                     void* __restrict__ Cv, _Float16* __restrict__ Qo,
                     _Float16* __restrict__ Ko, _Float16* __restrict__ Vt,
                     int M, int N, int K) {
  __shared__ _Float16 lA[2][MBLK * LDSTR];
  __shared__ _Float16 lB[2][NBLK * LDSTR];

  const int tid   = threadIdx.x;
  const int mBase = blockIdx.y * MBLK;
  const int nBase = blockIdx.x * NBLK;
  const int wave  = tid >> 5;
  const int lane  = tid & 31;
  const int r     = lane & 15, hh = lane >> 4;
  const int mw    = wave >> 1;             // 0..3 : 32-row band
  const int nw    = wave & 1;              // 0..1 : 64-col band

  // symmetric global->LDS staging: each thread stages 16 halves of A and of B
  const int sRow = tid >> 1;               // 0..127
  const int sCol = (tid & 1) << 4;         // 0 | 16 halves
  int aR = mBase + sRow; if (aR > M - 1) aR = M - 1;    // clamp (head GEMMs, M=16)
  const _Float16* aSrc = A + (size_t)aR * K + sCol;
  const _Float16* bSrc = W + (size_t)(nBase + sRow) * K + sCol;
  _Float16* laD[2] = { &lA[0][sRow * LDSTR + sCol], &lA[1][sRow * LDSTR + sCol] };
  _Float16* lbD[2] = { &lB[0][sRow * LDSTR + sCol], &lB[1][sRow * LDSTR + sCol] };

  v8f acc[2][4];
#pragma unroll
  for (int mi = 0; mi < 2; ++mi)
#pragma unroll
    for (int ni = 0; ni < 4; ++ni)
#pragma unroll
      for (int j = 0; j < 8; ++j) acc[mi][ni][j] = 0.0f;

  const int nk = K / KSTEP;
  float4 ra0 = *(const float4*)aSrc;
  float4 ra1 = *(const float4*)(aSrc + 8);
  float4 rb0 = *(const float4*)bSrc;
  float4 rb1 = *(const float4*)(bSrc + 8);
  *(float4*)laD[0] = ra0; *(float4*)(laD[0] + 8) = ra1;
  *(float4*)lbD[0] = rb0; *(float4*)(lbD[0] + 8) = rb1;
  __syncthreads();

  for (int k = 0; k < nk; ++k) {
    const int cur = k & 1;
    if (k + 1 < nk) {
      const _Float16* ap = aSrc + (k + 1) * KSTEP;
      const _Float16* bp = bSrc + (k + 1) * KSTEP;
      ra0 = *(const float4*)ap; ra1 = *(const float4*)(ap + 8);
      rb0 = *(const float4*)bp; rb1 = *(const float4*)(bp + 8);
    }
    const _Float16* Ab = lA[cur];
    const _Float16* Bb = lB[cur];
    v16h af[2], bf[4];
    af[0] = frag_lds(Ab, mw * 32 + r, hh);
    af[1] = frag_lds(Ab, mw * 32 + 16 + r, hh);
#pragma unroll
    for (int ni = 0; ni < 4; ++ni) bf[ni] = frag_lds(Bb, nw * 64 + ni * 16 + r, hh);
#pragma unroll
    for (int mi = 0; mi < 2; ++mi)
#pragma unroll
      for (int ni = 0; ni < 4; ++ni) acc[mi][ni] = WMMA16(af[mi], bf[ni], acc[mi][ni]);
    if (k + 1 < nk) {
      const int nxt = cur ^ 1;
      *(float4*)laD[nxt] = ra0; *(float4*)(laD[nxt] + 8) = ra1;
      *(float4*)lbD[nxt] = rb0; *(float4*)(lbD[nxt] + 8) = rb1;
    }
    __syncthreads();
  }

  // epilogue: C/D layout: lane holds col n=r, rows hh*8+j
  float bv[4];
#pragma unroll
  for (int ni = 0; ni < 4; ++ni) bv[ni] = bias[nBase + nw * 64 + ni * 16 + r];
#pragma unroll
  for (int mi = 0; mi < 2; ++mi)
#pragma unroll
    for (int ni = 0; ni < 4; ++ni) {
      const int mT = mBase + mw * 32 + mi * 16;
      const int nT = nBase + nw * 64 + ni * 16;
#pragma unroll
      for (int j = 0; j < 8; ++j) {
        const int row = mT + hh * 8 + j;
        if (row < M) {
          float v = acc[mi][ni][j] + bv[ni];
          if (FUSE == 1) v = gelu_exact(v);
          if (FUSE == 4) {
            const int seg = nT / DIMD;                 // uniform per ni-tile
            const int col = (nT - seg * DIMD) + r;     // 0..767
            if (seg == 0)      Qo[(size_t)row * DIMD + col] = (_Float16)v;
            else if (seg == 1) Ko[(size_t)row * DIMD + col] = (_Float16)v;
            else {
              const int hI = col >> 6, dd = col & 63;
              const int bI = row >> 8, tI = row & 255;
              Vt[((size_t)(bI * NHEAD + hI) * HD + dd) * SEQ + tI] = (_Float16)v;
            }
          } else {
            const size_t idx = (size_t)row * N + nT + r;
            if (FUSE == 2)      ((float*)Cv)[idx] = v + R[idx];
            else if (FUSE == 3) ((float*)Cv)[idx] = v;
            else                ((_Float16*)Cv)[idx] = (_Float16)v;
          }
        }
      }
    }
}

// ---------------------------------------------------------------------------
// Fused flash attention: one wave per (b, h, 16-query tile).
// Streams K/VT in 32-key steps: QK^T (4 WMMA) -> online softmax (intra-half
// shfl reductions) -> P transposed via per-wave private LDS back into
// A-fragment layout -> P.V (4 WMMA, contiguous b128 B-frags from VT).
// ---------------------------------------------------------------------------
__global__ __launch_bounds__(256)
void flash_attn_kernel(const _Float16* __restrict__ Q, const _Float16* __restrict__ Km,
                       const _Float16* __restrict__ VT, const float* __restrict__ pad,
                       _Float16* __restrict__ Y) {
  __shared__ _Float16 pbuf[8][16 * LDSTR];
  const int wv = blockIdx.x * (blockDim.x >> 5) + (threadIdx.x >> 5);
  if (wv >= NBATCH * NHEAD * 16) return;       // wave-uniform
  const int qT = wv & 15;
  const int bh = wv >> 4;
  const int h = bh % NHEAD, b = bh / NHEAD;
  const int lane = threadIdx.x & 31;
  const int r = lane & 15, hh = lane >> 4;
  _Float16* pb = pbuf[threadIdx.x >> 5];

  const _Float16* qp = Q + (size_t)(b * SEQ + qT * 16 + r) * DIMD + h * HD + hh * 8;
  const v16h qf0 = frag_gbl_h(qp);
  const v16h qf1 = frag_gbl_h(qp + 32);

  float vq[8];
#pragma unroll
  for (int j = 0; j < 8; ++j) vq[j] = 1.0f - pad[b * SEQ + qT * 16 + hh * 8 + j];

  v8f o[4];
#pragma unroll
  for (int t = 0; t < 4; ++t)
#pragma unroll
    for (int j = 0; j < 8; ++j) o[t][j] = 0.0f;
  float m_run[8], l_run[8];
#pragma unroll
  for (int j = 0; j < 8; ++j) { m_run[j] = -1e30f; l_run[j] = 0.0f; }

  for (int kb = 0; kb < SEQ / 32; ++kb) {
    const int kbase = kb * 32;
    float val[2][8];
#pragma unroll
    for (int s = 0; s < 2; ++s) {
      const _Float16* kp =
          Km + (size_t)(b * SEQ + kbase + s * 16 + r) * DIMD + h * HD + hh * 8;
      v8f sacc = {};
      sacc = WMMA16(qf0, frag_gbl_h(kp), sacc);
      sacc = WMMA16(qf1, frag_gbl_h(kp + 32), sacc);
      const float vk = 1.0f - pad[b * SEQ + kbase + s * 16 + r];
#pragma unroll
      for (int j = 0; j < 8; ++j)
        val[s][j] = (vq[j] * vk > 0.0f) ? sacc[j] * 0.125f : -1e9f;
    }
    float ex0[8], ex1[8], corr[8];
#pragma unroll
    for (int j = 0; j < 8; ++j) {
      float mt = fmaxf(val[0][j], val[1][j]);
#pragma unroll
      for (int off = 8; off > 0; off >>= 1) mt = fmaxf(mt, __shfl_xor(mt, off, 32));
      const float mnew = fmaxf(m_run[j], mt);
      ex0[j] = __expf(val[0][j] - mnew);
      ex1[j] = __expf(val[1][j] - mnew);
      float rs = ex0[j] + ex1[j];
#pragma unroll
      for (int off = 8; off > 0; off >>= 1) rs += __shfl_xor(rs, off, 32);
      corr[j] = __expf(m_run[j] - mnew);
      l_run[j] = l_run[j] * corr[j] + rs;
      m_run[j] = mnew;
    }
#pragma unroll
    for (int t = 0; t < 4; ++t)
#pragma unroll
      for (int j = 0; j < 8; ++j) o[t][j] *= corr[j];
    // transpose 16x32 P tile into A-fragment layout (per-wave LDS, in-order DS)
#pragma unroll
    for (int j = 0; j < 8; ++j) {
      pb[(hh * 8 + j) * LDSTR + r]      = (_Float16)ex0[j];
      pb[(hh * 8 + j) * LDSTR + 16 + r] = (_Float16)ex1[j];
    }
    const v16h pf = frag_lds(pb, r, hh);
#pragma unroll
    for (int t = 0; t < 4; ++t) {
      const _Float16* vp =
          VT + ((size_t)(b * NHEAD + h) * HD + t * 16 + r) * SEQ + kbase + hh * 8;
      o[t] = WMMA16(pf, frag_gbl_h(vp), o[t]);
    }
  }
#pragma unroll
  for (int j = 0; j < 8; ++j) l_run[j] = 1.0f / l_run[j];
#pragma unroll
  for (int t = 0; t < 4; ++t)
#pragma unroll
    for (int j = 0; j < 8; ++j)
      Y[(size_t)(b * SEQ + qT * 16 + hh * 8 + j) * DIMD + h * HD + t * 16 + r] =
          (_Float16)(o[t][j] * l_run[j]);
}

// LayerNorm over 768 cols, f32 in -> f16 out (feeds GEMMs only)
__global__ void layernorm_rows_kernel(const float* __restrict__ X, const float* __restrict__ s,
                                      const float* __restrict__ bb, _Float16* __restrict__ O,
                                      int rows) {
  const int row = blockIdx.x * (blockDim.x >> 5) + (threadIdx.x >> 5);
  if (row >= rows) return;
  const int lane = threadIdx.x & 31;
  const float* x = X + (size_t)row * DIMD;
  float v[24];
  float sum = 0.0f;
#pragma unroll
  for (int i = 0; i < 24; ++i) { v[i] = x[lane + 32 * i]; sum += v[i]; }
#pragma unroll
  for (int off = 16; off > 0; off >>= 1) sum += __shfl_xor(sum, off, 32);
  const float mean = sum * (1.0f / DIMD);
  float var = 0.0f;
#pragma unroll
  for (int i = 0; i < 24; ++i) { float d = v[i] - mean; var += d * d; }
#pragma unroll
  for (int off = 16; off > 0; off >>= 1) var += __shfl_xor(var, off, 32);
  const float rstd = rsqrtf(var * (1.0f / DIMD) + LN_EPS);
  _Float16* o = O + (size_t)row * DIMD;
#pragma unroll
  for (int i = 0; i < 24; ++i) {
    const int c = lane + 32 * i;
    o[c] = (_Float16)((v[i] - mean) * rstd * s[c] + bb[c]);
  }
}

__global__ void l2norm_rows_kernel(float* __restrict__ X, int rows) {
  const int row = blockIdx.x * (blockDim.x >> 5) + (threadIdx.x >> 5);
  if (row >= rows) return;
  const int lane = threadIdx.x & 31;
  float* x = X + (size_t)row * DIMD;
  float v[24];
  float ss = 0.0f;
#pragma unroll
  for (int i = 0; i < 24; ++i) { v[i] = x[lane + 32 * i]; ss += v[i] * v[i]; }
#pragma unroll
  for (int off = 16; off > 0; off >>= 1) ss += __shfl_xor(ss, off, 32);
  const float inv = rsqrtf(ss);
#pragma unroll
  for (int i = 0; i < 24; ++i) x[lane + 32 * i] = v[i] * inv;
}

// Masked mean pool
__global__ void pool_kernel(const float* __restrict__ X, const float* __restrict__ pad,
                            float* __restrict__ O) {
  const int b = blockIdx.x;
  float cnt = 0.0f;
  for (int t = 0; t < SEQ; ++t) cnt += 1.0f - pad[b * SEQ + t];
  const float inv = 1.0f / fmaxf(cnt, 1.0f);
  for (int d = threadIdx.x; d < DIMD; d += blockDim.x) {
    float s = 0.0f;
    for (int t = 0; t < SEQ; ++t)
      s += X[(size_t)(b * SEQ + t) * DIMD + d] * (1.0f - pad[b * SEQ + t]);
    O[b * DIMD + d] = s * inv;
  }
}

// Final 16x16 logits, K=768, single wave
__global__ void clip_logits_kernel(const float* __restrict__ FF, const float* __restrict__ FP,
                                   const float* __restrict__ ls, float* __restrict__ out) {
  const int lane = threadIdx.x & 31;
  const int r = lane & 15, hh = lane >> 4;
  const float* aRow = FF + (size_t)r * DIMD + hh * 8;
  const float* bRow = FP + (size_t)r * DIMD + hh * 8;
  v8f acc = {};
  for (int k0 = 0; k0 < DIMD; k0 += 32) {
    acc = WMMA16(frag_gbl_f32(aRow + k0), frag_gbl_f32(bRow + k0), acc);
  }
  const float scale = __expf(ls[0]);
#pragma unroll
  for (int j = 0; j < 8; ++j) out[(hh * 8 + j) * NBATCH + r] = acc[j] * scale;
}

// f32 -> f16 vectorized conversion (n multiple of 4)
__global__ void cvt4_kernel(const float* __restrict__ src, _Float16* __restrict__ dst, int n4) {
  const int i = blockIdx.x * blockDim.x + threadIdx.x;
  if (i >= n4) return;
  const float4 v = ((const float4*)src)[i];
  v4h h;
  h[0] = (_Float16)v.x; h[1] = (_Float16)v.y; h[2] = (_Float16)v.z; h[3] = (_Float16)v.w;
  ((v4h*)dst)[i] = h;
}

// Embeddings
__global__ void embed_formula_kernel(const float* __restrict__ seq, const float* __restrict__ tw,
                                     const float* __restrict__ tb, float* __restrict__ X) {
  const int idx = blockIdx.x * blockDim.x + threadIdx.x;
  if (idx >= MROWS * DIMD) return;
  const int d = idx % DIMD;
  const int bt = idx / DIMD;
  const int t = bt % SEQ;
  float v = seq[bt] * tw[d] + tb[d];
  const int p2 = d & ~1;
  const float freq = __expf(-(float)p2 * (9.210340371976184f / (float)DIMD));
  const float ang = (float)t * freq;
  v += (d & 1) ? cosf(ang) : sinf(ang);
  X[idx] = v;
}

__global__ void embed_points_kernel(const float* __restrict__ pts,
                                    const float* __restrict__ w1, const float* __restrict__ b1,
                                    const float* __restrict__ w2, const float* __restrict__ b2,
                                    const float* __restrict__ w3, const float* __restrict__ b3,
                                    float* __restrict__ X) {
  const int idx = blockIdx.x * blockDim.x + threadIdx.x;
  if (idx >= MROWS * DIMD) return;
  const int d = idx % DIMD;
  const int bn = idx / DIMD;
  const int n = bn % SEQ, b = bn / SEQ;
  const int c = d / (DIMD / 3);
  const int j = d - c * (DIMD / 3);
  const float* w  = (c == 0) ? w1 : (c == 1) ? w2 : w3;
  const float* bi = (c == 0) ? b1 : (c == 1) ? b2 : b3;
  X[idx] = pts[(size_t)(b * 3 + c) * SEQ + n] * w[j] + bi[j];
}

// ===========================================================================
// Host orchestration
// ===========================================================================
static inline void launch_gemm(int fuse, const _Float16* A, const _Float16* W,
                               const float* bias, const float* R, void* C,
                               int M, int N, int K, hipStream_t s,
                               _Float16* Qo = nullptr, _Float16* Ko = nullptr,
                               _Float16* Vt = nullptr) {
  dim3 grid(N / NBLK, (M + MBLK - 1) / MBLK);
  if (fuse == 0)      wmma_gemm_tiled<0><<<grid, 256, 0, s>>>(A, W, bias, R, C, Qo, Ko, Vt, M, N, K);
  else if (fuse == 1) wmma_gemm_tiled<1><<<grid, 256, 0, s>>>(A, W, bias, R, C, Qo, Ko, Vt, M, N, K);
  else if (fuse == 2) wmma_gemm_tiled<2><<<grid, 256, 0, s>>>(A, W, bias, R, C, Qo, Ko, Vt, M, N, K);
  else if (fuse == 3) wmma_gemm_tiled<3><<<grid, 256, 0, s>>>(A, W, bias, R, C, Qo, Ko, Vt, M, N, K);
  else                wmma_gemm_tiled<4><<<grid, 256, 0, s>>>(A, W, bias, R, C, Qo, Ko, Vt, M, N, K);
}

static inline void launch_cvt(const float* src, _Float16* dst, size_t n, hipStream_t s) {
  const int n4 = (int)(n / 4);
  cvt4_kernel<<<(n4 + 255) / 256, 256, 0, s>>>(src, dst, n4);
}

// P: 16 leaf pointers in _block_params insertion order.
// W16 packed per branch: [Wqkv (L x 3 x D x D, layer-interleaved) | Wo | Wm1 | Wm2]
static void run_encoder(void* const* P, const float* pad, _Float16* W16, float* BQKV,
                        float* X, _Float16* XLN, _Float16* Q16, _Float16* K16,
                        _Float16* VT16, _Float16* Y16, _Float16* H16, hipStream_t s) {
  const float* ln1_s = (const float*)P[0];
  const float* ln1_b = (const float*)P[1];
  const float* ln2_s = (const float*)P[2];
  const float* ln2_b = (const float*)P[3];
  const float* bq  = (const float*)P[5];
  const float* bk  = (const float*)P[7];
  const float* bv  = (const float*)P[9];
  const float* bo  = (const float*)P[11];
  const float* bm1 = (const float*)P[13];
  const float* bm2 = (const float*)P[15];

  const size_t DD = (size_t)DIMD * DIMD;
  const size_t DM = (size_t)FFDIM * DIMD;
  _Float16* Wqkv = W16;                         // L * 3 * DD
  _Float16* Wo16 = Wqkv + (size_t)NLAYER * 3 * DD;
  _Float16* Wm1h = Wo16 + NLAYER * DD;
  _Float16* Wm2h = Wm1h + NLAYER * DM;

  // pack Wq/Wk/Wv per layer into [Wq_i | Wk_i | Wv_i] + pack f32 biases
  for (int i = 0; i < NLAYER; ++i) {
    launch_cvt((const float*)P[4] + i * DD, Wqkv + (size_t)(3 * i + 0) * DD, DD, s);
    launch_cvt((const float*)P[6] + i * DD, Wqkv + (size_t)(3 * i + 1) * DD, DD, s);
    launch_cvt((const float*)P[8] + i * DD, Wqkv + (size_t)(3 * i + 2) * DD, DD, s);
    hipMemcpyAsync(BQKV + i * QKVN,            bq + i * DIMD, DIMD * 4, hipMemcpyDeviceToDevice, s);
    hipMemcpyAsync(BQKV + i * QKVN + DIMD,     bk + i * DIMD, DIMD * 4, hipMemcpyDeviceToDevice, s);
    hipMemcpyAsync(BQKV + i * QKVN + 2 * DIMD, bv + i * DIMD, DIMD * 4, hipMemcpyDeviceToDevice, s);
  }
  launch_cvt((const float*)P[10], Wo16, NLAYER * DD, s);
  launch_cvt((const float*)P[12], Wm1h, NLAYER * DM, s);
  launch_cvt((const float*)P[14], Wm2h, NLAYER * DM, s);

  const int lnBlocks = (MROWS + 7) / 8;
  const int faBlocks = (NBATCH * NHEAD * 16) / 8;   // 384

  for (int i = 0; i < NLAYER; ++i) {
    layernorm_rows_kernel<<<lnBlocks, 256, 0, s>>>(X, ln1_s + i * DIMD, ln1_b + i * DIMD,
                                                   XLN, MROWS);
    launch_gemm(4, XLN, Wqkv + (size_t)(3 * i) * DD, BQKV + i * QKVN, nullptr, nullptr,
                MROWS, QKVN, DIMD, s, Q16, K16, VT16);
    flash_attn_kernel<<<faBlocks, 256, 0, s>>>(Q16, K16, VT16, pad, Y16);
    launch_gemm(2, Y16, Wo16 + i * DD, bo + i * DIMD, X, X, MROWS, DIMD, DIMD, s);
    layernorm_rows_kernel<<<lnBlocks, 256, 0, s>>>(X, ln2_s + i * DIMD, ln2_b + i * DIMD,
                                                   XLN, MROWS);
    launch_gemm(1, XLN, Wm1h + i * DM, bm1 + i * FFDIM, nullptr, H16, MROWS, FFDIM, DIMD, s);
    launch_gemm(2, H16, Wm2h + i * DM, bm2 + i * DIMD, X, X, MROWS, DIMD, FFDIM, s);
  }
}

extern "C" void kernel_launch(void* const* d_in, const int* in_sizes, int n_in,
                              void* d_out, int out_size, void* d_ws, size_t ws_size,
                              hipStream_t stream) {
  (void)in_sizes; (void)n_in; (void)out_size; (void)ws_size;
  const float* fseq = (const float*)d_in[0];
  const float* fpad = (const float*)d_in[1];
  const float* pts  = (const float*)d_in[2];
  const float* ppad = (const float*)d_in[3];
  void* const* fP = d_in + 4;
  void* const* pP = d_in + 20;
  const float* tok_w = (const float*)d_in[36];
  const float* tok_b = (const float*)d_in[37];
  const float* p1w = (const float*)d_in[38]; const float* p1b = (const float*)d_in[39];
  const float* p2w = (const float*)d_in[40]; const float* p2b = (const float*)d_in[41];
  const float* p3w = (const float*)d_in[42]; const float* p3b = (const float*)d_in[43];
  const float* lnf_s = (const float*)d_in[44]; const float* lnf_b = (const float*)d_in[45];
  const float* lnp_s = (const float*)d_in[46]; const float* lnp_b = (const float*)d_in[47];
  const float* pf_W1 = (const float*)d_in[48]; const float* pf_b1 = (const float*)d_in[49];
  const float* pf_W2 = (const float*)d_in[50]; const float* pf_b2 = (const float*)d_in[51];
  const float* pp_W1 = (const float*)d_in[52]; const float* pp_b1 = (const float*)d_in[53];
  const float* pp_W2 = (const float*)d_in[54]; const float* pp_b2 = (const float*)d_in[55];
  const float* lscale = (const float*)d_in[56];

  // ---- workspace layout (float units; halves packed 2-per-float) ----
  const size_t ACT  = (size_t)MROWS * DIMD;                  // 3,145,728
  const size_t ACTH = ACT / 2;
  const size_t HID  = (size_t)MROWS * FFDIM;                 // halves
  const size_t WENC_H = (size_t)NLAYER * 4 * DIMD * DIMD + 2 * (size_t)NLAYER * FFDIM * DIMD;
  const size_t PROJ = (size_t)2 * DIMD * DIMD;               // one head fc leaf

  float* ws = (float*)d_ws;
  float*     X    = ws;                 ws += ACT;
  _Float16*  XLN  = (_Float16*)ws;      ws += ACTH;
  _Float16*  Q16  = (_Float16*)ws;      ws += ACTH;
  _Float16*  K16  = (_Float16*)ws;      ws += ACTH;
  _Float16*  VT16 = (_Float16*)ws;      ws += ACTH;          // V transposed [b,h,d,t]
  _Float16*  Y16  = (_Float16*)ws;      ws += ACTH;
  _Float16*  H16  = (_Float16*)ws;      ws += HID / 2;       // MLP hidden (f16)
  _Float16*  W16  = (_Float16*)ws;      ws += WENC_H / 2;    // per-branch f16 weights
  _Float16*  HW16 = (_Float16*)ws;      ws += (4 * PROJ) / 2;
  float*     BQKV = ws;                 ws += NLAYER * QKVN; // packed qkv bias (f32)
  float*     POOLF = ws;                ws += NBATCH * DIMD;
  float*     POOLP = ws;                ws += NBATCH * DIMD;
  _Float16*  LNH16 = (_Float16*)ws;     ws += (NBATCH * DIMD) / 2;
  _Float16*  HMID  = (_Float16*)ws;     ws += (NBATCH * 2 * DIMD) / 2;
  float*     FF    = ws;                ws += NBATCH * DIMD;
  float*     FP    = ws;                ws += NBATCH * DIMD;

  _Float16* pfW1h = HW16;
  _Float16* pfW2h = HW16 + PROJ;
  _Float16* ppW1h = HW16 + 2 * PROJ;
  _Float16* ppW2h = HW16 + 3 * PROJ;

  const int embBlocks = (MROWS * DIMD + 255) / 256;

  // ---- formula branch ----
  embed_formula_kernel<<<embBlocks, 256, 0, stream>>>(fseq, tok_w, tok_b, X);
  run_encoder(fP, fpad, W16, BQKV, X, XLN, Q16, K16, VT16, Y16, H16, stream);
  pool_kernel<<<NBATCH, 256, 0, stream>>>(X, fpad, POOLF);

  // ---- points branch (reuses every buffer incl. W16) ----
  embed_points_kernel<<<embBlocks, 256, 0, stream>>>(pts, p1w, p1b, p2w, p2b, p3w, p3b, X);
  run_encoder(pP, ppad, W16, BQKV, X, XLN, Q16, K16, VT16, Y16, H16, stream);
  pool_kernel<<<NBATCH, 256, 0, stream>>>(X, ppad, POOLP);

  // ---- heads ----
  launch_cvt(pf_W1, pfW1h, PROJ, stream);
  launch_cvt(pf_W2, pfW2h, PROJ, stream);
  launch_cvt(pp_W1, ppW1h, PROJ, stream);
  launch_cvt(pp_W2, ppW2h, PROJ, stream);

  layernorm_rows_kernel<<<(NBATCH + 7) / 8, 256, 0, stream>>>(POOLF, lnf_s, lnf_b, LNH16, NBATCH);
  launch_gemm(1, LNH16, pfW1h, pf_b1, nullptr, HMID, NBATCH, 2 * DIMD, DIMD, stream);
  launch_gemm(3, HMID, pfW2h, pf_b2, nullptr, FF, NBATCH, DIMD, 2 * DIMD, stream);

  layernorm_rows_kernel<<<(NBATCH + 7) / 8, 256, 0, stream>>>(POOLP, lnp_s, lnp_b, LNH16, NBATCH);
  launch_gemm(1, LNH16, ppW1h, pp_b1, nullptr, HMID, NBATCH, 2 * DIMD, DIMD, stream);
  launch_gemm(3, HMID, ppW2h, pp_b2, nullptr, FP, NBATCH, DIMD, 2 * DIMD, stream);

  l2norm_rows_kernel<<<(NBATCH + 7) / 8, 256, 0, stream>>>(FF, NBATCH);
  l2norm_rows_kernel<<<(NBATCH + 7) / 8, 256, 0, stream>>>(FP, NBATCH);

  clip_logits_kernel<<<1, 32, 0, stream>>>(FF, FP, lscale, (float*)d_out);
}
